// CachedMultiHeadAttention_43731357008608
// MI455X (gfx1250) — compile-verified, matrix operand order checked by reading, not run on hardware
//
#include <hip/hip_runtime.h>
#include <hip/hip_bf16.h>

// ---- problem constants (match reference) ----
#define BB        8
#define QQ        128
#define DD        2048
#define HH        16
#define HDIM      128
#define CACHE_LEN 3968
#define KLEN      4096   // CACHE + Q

typedef __bf16 bf16_t;
typedef __attribute__((ext_vector_type(16))) __bf16 v16bf;
typedef __attribute__((ext_vector_type(8)))  __bf16 v8bf;
typedef __attribute__((ext_vector_type(8)))  float  v8f;
typedef __attribute__((ext_vector_type(4)))  float  v4f;

union bfvec {           // one WMMA A/B operand (16 bf16 = 8 VGPRs)
  v16bf    v;
  v8bf     h[2];
  unsigned u[8];
};

// two f32 -> packed bf16x2 in one v_perm_b32 (truncation)
static __device__ __forceinline__ unsigned pack_bf16x2(float lo, float hi) {
  union { float f; unsigned u; } a, b;
  a.f = lo; b.f = hi;
  return __builtin_amdgcn_perm(b.u, a.u, 0x07060302u);
}

static __device__ __forceinline__ bf16_t f2bf(float f) {
  union { float f; unsigned u; } v; v.f = f;
  unsigned short s = (unsigned short)(v.u >> 16);
  bf16_t b;
  __builtin_memcpy(&b, &s, sizeof(b));
  return b;
}

static __device__ __forceinline__ v8f wmma_bf16(v16bf a, v16bf b, v8f c) {
  return __builtin_amdgcn_wmma_f32_16x16x32_bf16(false, a, false, b, (short)0, c,
                                                 false, false);
}

// =====================================================================
// Converters (one-time passes).
// =====================================================================
__global__ __launch_bounds__(256)
void cvt_bf16_kernel(const float* __restrict__ src, unsigned* __restrict__ dst,
                     int nvec) {
  const int i = blockIdx.x * 256 + threadIdx.x;
  if (i < nvec) {
    const v4f f = ((const v4f*)src)[i];
    uint2 o;
    o.x = pack_bf16x2(f[0], f[1]);
    o.y = pack_bf16x2(f[2], f[3]);
    ((uint2*)dst)[i] = o;
  }
}

// f32 W[r][c] (r<rows, c<cols) -> bf16 Wt[c][r]  (reads coalesced on c)
__global__ __launch_bounds__(256)
void cvt_transpose_kernel(const float* __restrict__ src, bf16_t* __restrict__ dst,
                          int rows, int cols) {
  const int c = blockIdx.x * 256 + threadIdx.x;
  const int r = blockIdx.y;
  dst[(size_t)c * rows + r] = f2bf(src[(size_t)r * cols + c]);
}

// =====================================================================
// Kernel 1: QKV projection.  Y[1024 x 6144] = xb @ WqkvT^T (both bf16).
// One wave computes a 16x64 tile (4 accumulators, A reused 4x).
// =====================================================================
__global__ __launch_bounds__(32)
void qkv_gemm_kernel(const bf16_t* __restrict__ xb, const bf16_t* __restrict__ Wt,
                     float* __restrict__ qo, float* __restrict__ ko,
                     float* __restrict__ vo) {
  const int n0   = blockIdx.x * 64;
  const int m0   = blockIdx.y * 16;
  const int lane = threadIdx.x;
  const int half = lane >> 4;
  const int l    = lane & 15;

  const v8f vzero = {};
  v8f acc[4];
#pragma unroll
  for (int t = 0; t < 4; ++t) acc[t] = vzero;

  const bf16_t* arow = xb + (size_t)(m0 + l) * DD;
  const bf16_t* brow[4];
#pragma unroll
  for (int t = 0; t < 4; ++t) brow[t] = Wt + (size_t)(n0 + t * 16 + l) * DD;

  for (int k0 = 0; k0 < DD; k0 += 32) {
    const int ka = k0 + half * 8;
    bfvec A;
    A.h[0] = *(const v8bf*)(arow + ka);
    A.h[1] = *(const v8bf*)(arow + ka + 16);
#pragma unroll
    for (int t = 0; t < 4; ++t) {
      bfvec Bv;
      Bv.h[0] = *(const v8bf*)(brow[t] + ka);
      Bv.h[1] = *(const v8bf*)(brow[t] + ka + 16);
      acc[t] = wmma_bf16(A.v, Bv.v, acc[t]);
    }
  }

  // scatter into q/k/v [B,H,Q,HD]; rows m0..m0+15 share one batch index,
  // so hoist the base address and stride by HDIM per accumulator row.
  const int bidx = m0 >> 7;
  const int q0   = (m0 & 127) + 8 * half;
#pragma unroll
  for (int t = 0; t < 4; ++t) {
    const int n = n0 + t * 16 + l;
    const int s = n >> 11, rem = n & 2047;
    const int hh = rem >> 7, hd = rem & 127;
    float* dst = (s == 0) ? qo : (s == 1) ? ko : vo;
    float* base = dst + (((size_t)bidx * HH + hh) * QQ + q0) * HDIM + hd;
#pragma unroll
    for (int r = 0; r < 8; ++r)
      base[r * HDIM] = acc[t][r];
  }
}

// =====================================================================
// Kernel 2: flash attention. One 256-thread block (8 waves) per (b,h);
// wave w owns query tile w.  Double-buffered LDS staging: the NEXT 32-key
// K/V block's global loads are issued into registers before computing the
// current block, and only waited/packed/stored after the WMMAs -> global
// latency overlaps compute even at 1 workgroup/WGP.  KV read exactly once.
// =====================================================================
__global__ __launch_bounds__(256)
void attn_kernel(const float* __restrict__ kc, const float* __restrict__ vc,
                 const float* __restrict__ qn, const float* __restrict__ kn,
                 const float* __restrict__ vn, bf16_t* __restrict__ attnb) {
  __shared__ __attribute__((aligned(16))) bf16_t Kt[2][32 * HDIM];   // [key][hd]
  __shared__ __attribute__((aligned(16))) bf16_t Vt[2][HDIM * 32];   // [hd][key]
  __shared__ __attribute__((aligned(16))) bf16_t Pt[8 * 16 * 32];    // per-wave P

  const int bh = blockIdx.x;            // [0,128)
  const int h  = bh & 15;
  const int b  = bh >> 4;

  const int tid  = threadIdx.x;
  const int w    = tid >> 5;            // wave id -> query tile
  const int lane = tid & 31;
  const int half = lane >> 4;
  const int l    = lane & 15;
  const int qb   = w * 16;

  const float scale = 0.08838834764831844f;   // 1/sqrt(128)

  // ---- per-wave Q fragments (scaled, f32 -> bf16 via v_perm) ----
  bfvec qa[4];
  {
    const float* qrow = qn + (((size_t)b * HH + h) * QQ + qb + l) * HDIM;
#pragma unroll
    for (int c = 0; c < 4; ++c) {
      const int base = c * 32 + half * 8;
      const v4f f0 = *(const v4f*)(qrow + base);
      const v4f f1 = *(const v4f*)(qrow + base + 4);
      const v4f f2 = *(const v4f*)(qrow + base + 16);
      const v4f f3 = *(const v4f*)(qrow + base + 20);
      qa[c].u[0] = pack_bf16x2(f0[0] * scale, f0[1] * scale);
      qa[c].u[1] = pack_bf16x2(f0[2] * scale, f0[3] * scale);
      qa[c].u[2] = pack_bf16x2(f1[0] * scale, f1[1] * scale);
      qa[c].u[3] = pack_bf16x2(f1[2] * scale, f1[3] * scale);
      qa[c].u[4] = pack_bf16x2(f2[0] * scale, f2[1] * scale);
      qa[c].u[5] = pack_bf16x2(f2[2] * scale, f2[3] * scale);
      qa[c].u[6] = pack_bf16x2(f3[0] * scale, f3[1] * scale);
      qa[c].u[7] = pack_bf16x2(f3[2] * scale, f3[3] * scale);
    }
  }

  const v8f vzero = {};
  v8f o[8];
#pragma unroll
  for (int t = 0; t < 8; ++t) o[t] = vzero;
  float mrun[8], lrun[8];
#pragma unroll
  for (int r = 0; r < 8; ++r) { mrun[r] = -__builtin_inff(); lrun[r] = 0.f; }

  const float* kcb = kc + ((size_t)b * HH + h) * CACHE_LEN * HDIM;
  const float* vcb = vc + ((size_t)b * HH + h) * CACHE_LEN * HDIM;
  const float* knb = kn + ((size_t)b * HH + h) * QQ * HDIM;
  const float* vnb = vn + ((size_t)b * HH + h) * QQ * HDIM;

  bf16_t* pW = Pt + w * (16 * 32);

  // in-flight staging registers for one 32-key K/V block
  v4f   kreg[4];
  float vreg[16];

  // staging geometry per thread (constant across iterations)
  const int kchunk_key[4] = { (tid + 0) >> 5, (tid + 256) >> 5,
                              (tid + 512) >> 5, (tid + 768) >> 5 };
  const int kchunk_hdc    = (tid & 31) << 2;
  const int v_hd          = tid & 127;
  const int v_kg          = tid >> 7;   // 0 or 1 -> plus 2*j below

  // issue: start global loads for key block kb (uniform cache/new select)
  auto issue = [&](int kb) {
    const bool incache = (kb < CACHE_LEN);   // 32 | CACHE_LEN
    const float* Kb = incache ? (kcb + (size_t)kb * HDIM)
                              : (knb + (size_t)(kb - CACHE_LEN) * HDIM);
    const float* Vb = incache ? (vcb + (size_t)kb * HDIM)
                              : (vnb + (size_t)(kb - CACHE_LEN) * HDIM);
#pragma unroll
    for (int j = 0; j < 4; ++j)
      kreg[j] = __builtin_nontemporal_load(
          (const v4f*)(Kb + (size_t)kchunk_key[j] * HDIM + kchunk_hdc));
#pragma unroll
    for (int j = 0; j < 4; ++j) {
      const int kg = v_kg + 2 * j;            // 0..7
#pragma unroll
      for (int jj = 0; jj < 4; ++jj)
        vreg[j * 4 + jj] = __builtin_nontemporal_load(
            Vb + (size_t)(kg * 4 + jj) * HDIM + v_hd);
    }
  };

  // commit: wait on those loads, pack f32->bf16, store into LDS buffer
  auto commit = [&](int buf) {
#pragma unroll
    for (int j = 0; j < 4; ++j) {
      uint2 oo;
      oo.x = pack_bf16x2(kreg[j][0], kreg[j][1]);
      oo.y = pack_bf16x2(kreg[j][2], kreg[j][3]);
      *(uint2*)&Kt[buf][kchunk_key[j] * HDIM + kchunk_hdc] = oo;
    }
#pragma unroll
    for (int j = 0; j < 4; ++j) {
      const int kg = v_kg + 2 * j;
      uint2 oo;
      oo.x = pack_bf16x2(vreg[j * 4 + 0], vreg[j * 4 + 1]);
      oo.y = pack_bf16x2(vreg[j * 4 + 2], vreg[j * 4 + 3]);
      *(uint2*)&Vt[buf][v_hd * 32 + kg * 4] = oo;
    }
  };

  // prologue: stage block 0 into buffer 0
  issue(0);
  commit(0);
  __syncthreads();

  int buf = 0;
  for (int kb = 0; kb < KLEN; kb += 32) {
    const bool more = (kb + 32 < KLEN);      // uniform
    if (more) issue(kb + 32);                // loads in flight during compute

    const bf16_t* Kc = Kt[buf];
    const bf16_t* Vc = Vt[buf];

    // ---- scores: S[16 x 32] = Q * K^T (K fragments from LDS) ----
    v8f s0 = vzero, s1 = vzero;
#pragma unroll
    for (int c = 0; c < 4; ++c) {
      const int base = c * 32 + half * 8;
      bfvec b0, b1;
      b0.h[0] = *(const v8bf*)&Kc[l * HDIM + base];
      b0.h[1] = *(const v8bf*)&Kc[l * HDIM + base + 16];
      b1.h[0] = *(const v8bf*)&Kc[(16 + l) * HDIM + base];
      b1.h[1] = *(const v8bf*)&Kc[(16 + l) * HDIM + base + 16];
      s0 = wmma_bf16(qa[c].v, b0.v, s0);
      s1 = wmma_bf16(qa[c].v, b1.v, s1);
    }

    // ---- causal mask + online softmax ----
    const int j0 = kb + l;
    const int j1 = j0 + 16;
#pragma unroll
    for (int r = 0; r < 8; ++r) {
      const int qi  = qb + r + 8 * half;
      const int lim = qi + (KLEN - QQ);
      float v0 = (j0 > lim) ? -__builtin_inff() : s0[r];
      float v1 = (j1 > lim) ? -__builtin_inff() : s1[r];
      float mx = fmaxf(v0, v1);
#pragma unroll
      for (int off = 8; off >= 1; off >>= 1)
        mx = fmaxf(mx, __shfl_xor(mx, off, 32));
      const float mnew = fmaxf(mrun[r], mx);
      const float corr = __expf(mrun[r] - mnew);
      const float p0 = __expf(v0 - mnew);
      const float p1 = __expf(v1 - mnew);
      float ps = p0 + p1;
#pragma unroll
      for (int off = 8; off >= 1; off >>= 1)
        ps += __shfl_xor(ps, off, 32);
      lrun[r] = lrun[r] * corr + ps;
      mrun[r] = mnew;
#pragma unroll
      for (int t = 0; t < 8; ++t) o[t][r] *= corr;
      const int row = r + 8 * half;
      pW[row * 32 + l]      = f2bf(p0);        // wave-private region:
      pW[row * 32 + 16 + l] = f2bf(p1);        // DS in-order per wave
    }

    // ---- P in A-operand layout, then O += P @ V ----
    bfvec pa;
    pa.h[0] = *(const v8bf*)&pW[l * 32 + half * 8];
    pa.h[1] = *(const v8bf*)&pW[l * 32 + 16 + half * 8];
#pragma unroll
    for (int t = 0; t < 8; ++t) {
      const int hd = t * 16 + l;
      bfvec vb;
      vb.h[0] = *(const v8bf*)&Vc[hd * 32 + half * 8];
      vb.h[1] = *(const v8bf*)&Vc[hd * 32 + 16 + half * 8];
      o[t] = wmma_bf16(pa.v, vb.v, o[t]);
    }

    // now (and only now) drain the in-flight loads into the other buffer
    if (more) commit(buf ^ 1);
    __syncthreads();
    buf ^= 1;
  }

  // ---- normalize, store bf16 activations [B, Q, H*HD] for the proj GEMM ----
#pragma unroll
  for (int r = 0; r < 8; ++r) {
    const int row = r + 8 * half;
    const float inv = 1.0f / lrun[r];
    bf16_t* dst = attnb + ((size_t)b * QQ + qb + row) * DD + h * HDIM;
#pragma unroll
    for (int t = 0; t < 8; ++t)
      dst[t * 16 + l] = f2bf(o[t][r] * inv);
  }
}

// =====================================================================
// Kernel 3: output projection.  out[1024 x 2048] = attnb @ WprojT^T + b.
// =====================================================================
__global__ __launch_bounds__(32)
void proj_gemm_kernel(const bf16_t* __restrict__ A, const bf16_t* __restrict__ Wt,
                      const float* __restrict__ bias, float* __restrict__ out) {
  const int n0   = blockIdx.x * 64;
  const int m0   = blockIdx.y * 16;
  const int lane = threadIdx.x;
  const int half = lane >> 4;
  const int l    = lane & 15;

  const v8f vzero = {};
  v8f acc[4];
#pragma unroll
  for (int t = 0; t < 4; ++t) acc[t] = vzero;

  const bf16_t* arow = A + (size_t)(m0 + l) * DD;
  const bf16_t* brow[4];
#pragma unroll
  for (int t = 0; t < 4; ++t) brow[t] = Wt + (size_t)(n0 + t * 16 + l) * DD;

  for (int k0 = 0; k0 < DD; k0 += 32) {
    const int ka = k0 + half * 8;
    bfvec Av;
    Av.h[0] = *(const v8bf*)(arow + ka);
    Av.h[1] = *(const v8bf*)(arow + ka + 16);
#pragma unroll
    for (int t = 0; t < 4; ++t) {
      bfvec Bv;
      Bv.h[0] = *(const v8bf*)(brow[t] + ka);
      Bv.h[1] = *(const v8bf*)(brow[t] + ka + 16);
      acc[t] = wmma_bf16(Av.v, Bv.v, acc[t]);
    }
  }

  float* base0 = out + (size_t)(m0 + 8 * half) * DD + n0 + l;
#pragma unroll
  for (int t = 0; t < 4; ++t) {
    const float bn = bias[n0 + t * 16 + l];
#pragma unroll
    for (int r = 0; r < 8; ++r)
      base0[(size_t)r * DD + t * 16] = acc[t][r] + bn;
  }
}

// =====================================================================
extern "C" void kernel_launch(void* const* d_in, const int* in_sizes, int n_in,
                              void* d_out, int out_size, void* d_ws, size_t ws_size,
                              hipStream_t stream) {
  const float* x     = (const float*)d_in[0];
  const float* kc    = (const float*)d_in[1];
  const float* vc    = (const float*)d_in[2];
  const float* Wqkv  = (const float*)d_in[3];
  const float* Wproj = (const float*)d_in[4];
  const float* bproj = (const float*)d_in[5];
  float* out = (float*)d_out;

  // ---- workspace carve-up (all chunks 16B-aligned) ----
  const size_t PER = (size_t)BB * HH * QQ * HDIM;  // 2,097,152 elements
  char* p = (char*)d_ws;
  float*  qn    = (float*)p;              p += PER * 4;           //  8 MB
  float*  kn    = (float*)p;              p += PER * 4;           //  8 MB
  float*  vn    = (float*)p;              p += PER * 4;           //  8 MB
  bf16_t* attnb = (bf16_t*)p;             p += PER * 2;           //  4 MB
  bf16_t* xb    = (bf16_t*)p;             p += PER * 2;           //  4 MB
  bf16_t* Wqkvt = (bf16_t*)p;             p += (size_t)DD * 3 * DD * 2; // 24 MB
  bf16_t* Wpt   = (bf16_t*)p;             p += (size_t)DD * DD * 2;     //  8 MB

  // ---- one-time conversions ----
  cvt_bf16_kernel<<<dim3((unsigned)((PER / 4 + 255) / 256)), dim3(256), 0, stream>>>(
      x, (unsigned*)xb, (int)(PER / 4));
  cvt_transpose_kernel<<<dim3(3 * DD / 256, DD), dim3(256), 0, stream>>>(
      Wqkv, Wqkvt, DD, 3 * DD);
  cvt_transpose_kernel<<<dim3(DD / 256, DD), dim3(256), 0, stream>>>(
      Wproj, Wpt, DD, DD);

  // ---- pipeline ----
  qkv_gemm_kernel<<<dim3(96, 64), dim3(32), 0, stream>>>(xb, Wqkvt, qn, kn, vn);
  attn_kernel<<<dim3(128), dim3(256), 0, stream>>>(kc, vc, qn, kn, vn, attnb);
  proj_gemm_kernel<<<dim3(32, 64), dim3(32), 0, stream>>>(attnb, Wpt, bproj, out);
}
